// WallConvNet_17721035063807
// MI455X (gfx1250) — matrix operand order repaired and checked by reading, not compiled, not run on hardware
//
#include <hip/hip_runtime.h>
#include <hip/hip_bf16.h>

typedef __attribute__((ext_vector_type(16))) _Float16 v16h;
typedef __attribute__((ext_vector_type(8)))  _Float16 v8h;
typedef __attribute__((ext_vector_type(8)))  float    v8f;

#define PADIDX 37

// ---- dynamic LDS arena (191360 B, WGP has 320KB) --------------------------
// Phase A:
//   img   : 0      .. 4608     (16*18 f32, zero-padded 14x16 hand image)
//   x1    : 4608   .. 23040    (32*16*18 f16, conv1 relu output w/ halo)
//   w2h   : 23040  .. 59904    (64*288 f16 conv2 weights)
//   X2f   : 59904  .. 188928   (9*224*32 f16 im2col in B-fragment layout)
// Phase B (reuses [0..23040)):
//   dkv   : 0      .. 10752    (96*28 f32)
//   dmask : 10752  .. 11136    (96 f32)
//   hemb  : 11136  .. 12032    (14*16 f32)
//   q     : 12032  .. 13600    (14*28 f32)
//   wlog  : 13600  .. 18976    (14*96 f32)
//   dq    : 18976  .. 20544    (14*28 f32)
//   pooled: 20544  .. 20800    (4*16 f32)
//   cf    : 20800  .. 21824    (4*64 f32)
// Persistent:
//   handfeat 188928 (64 f32), callfeat 189184 (64 f32),
//   dfeat    189440 (64 f32), h1s     189696 (128 f32),
//   ctab     190208 (288 i32) -> 191360
#define OFF_IMG   0
#define OFF_X1    4608
#define OFF_W2H   23040
#define OFF_X2F   59904
#define OFF_HF    188928
#define OFF_CF    189184
#define OFF_DF    189440
#define OFF_H1    189696
#define OFF_CTAB  190208
#define SMEM_TOTAL 191360

__global__ __launch_bounds__(256)
void wallconv_fused_kernel(const int* __restrict__ hand,      // (B,14)
                           const int* __restrict__ called,    // (B,16)
                           const int* __restrict__ opp,       // (B,3,16)
                           const int* __restrict__ disc,      // (B,4,24)
                           const float* __restrict__ temb,    // (38,16)
                           const float* __restrict__ c1w,     // (32,1,3,3)
                           const float* __restrict__ c1b,     // (32)
                           const float* __restrict__ c2w,     // (64,32,3,3)
                           const float* __restrict__ c2b,     // (64)
                           const float* __restrict__ clw,     // (64,16)
                           const float* __restrict__ clb,     // (64)
                           const float* __restrict__ crw,     // (16,64)
                           const float* __restrict__ crb,     // (16)
                           const float* __restrict__ pe,      // (24,8)
                           const float* __restrict__ ye,      // (4,4)
                           const float* __restrict__ qpw,     // (28,16)
                           const float* __restrict__ qpb,     // (28)
                           const float* __restrict__ qrw,     // (64,392)
                           const float* __restrict__ h1w,     // (128,192)
                           const float* __restrict__ h1b,     // (128)
                           const float* __restrict__ h2w,     // (37,128)
                           const float* __restrict__ h2b,     // (37)
                           float* __restrict__ out)           // (B,37)
{
    extern __shared__ __align__(16) char smem[];
    const int b   = blockIdx.x;
    const int tid = threadIdx.x;
    const int lane = tid & 31;
    const int wv   = tid >> 5;
    const int lg   = (lane < 16) ? 0 : 1;   // lane half-group
    const int ln   = lane & 15;

    float*    img      = (float*)(smem + OFF_IMG);
    _Float16* x1       = (_Float16*)(smem + OFF_X1);
    _Float16* w2h      = (_Float16*)(smem + OFF_W2H);
    _Float16* X2f      = (_Float16*)(smem + OFF_X2F);
    float*    handfeat = (float*)(smem + OFF_HF);
    float*    callfeat = (float*)(smem + OFF_CF);
    float*    dfeat    = (float*)(smem + OFF_DF);
    float*    h1s      = (float*)(smem + OFF_H1);
    int*      ctab     = (int*)(smem + OFF_CTAB);

    // pull conv2 weights toward cache early (global_prefetch_b8 path)
    __builtin_prefetch(c2w + tid * 72, 0, 1);

    // ---------------- Phase A: embed + conv1 + im2col + conv2 (WMMA) -------
    for (int i = tid; i < 16*18; i += 256) img[i] = 0.f;
    for (int i = tid; i < 32*16*18; i += 256) x1[i] = (_Float16)0.f;
    if (tid < 64) handfeat[tid] = 0.f;
    // im2col constant-offset table: K -> ci*288 + dy*18 + dx
    for (int k = tid; k < 288; k += 256) {
        int ci = k / 9, r = k - ci*9;
        ctab[k] = ci*288 + (r/3)*18 + (r - (r/3)*3);
    }
    __syncthreads();

    // gather hand embedding into padded image
    for (int i = tid; i < 224; i += 256) {
        int r = i >> 4, c = i & 15;
        int ix = hand[b*14 + r]; if (ix < 0) ix = PADIDX;
        img[(r+1)*18 + (c+1)] = temb[ix*16 + c];
    }
    // stage conv2 weights in LDS as f16 ([M=64][K=288], K = ci*9+dy*3+dx)
    for (int i = tid; i < 64*288; i += 256) w2h[i] = (_Float16)c2w[i];
    __syncthreads();

    // conv1 + relu -> x1 (f16, with halo preserved as zero)
    for (int o = tid; o < 32*224; o += 256) {
        int c = o / 224, p = o - c*224;
        int y = p >> 4, x = p & 15;
        float s = c1b[c];
        #pragma unroll
        for (int dy = 0; dy < 3; ++dy)
            #pragma unroll
            for (int dx = 0; dx < 3; ++dx)
                s += img[(y+dy)*18 + (x+dx)] * c1w[c*9 + dy*3 + dx];
        s = s > 0.f ? s : 0.f;
        x1[(c*16 + (y+1))*18 + (x+1)] = (_Float16)s;
    }
    __syncthreads();

    // build im2col in B-fragment layout: X2f[(kk*224+n)*32 + k32]
    // each thread owns a fixed pair of K-slots (2*k16, 2*k16+1) and walks (kk,n)
    {
        const int k16 = tid & 15;
        for (int m = tid >> 4; m < 9*224; m += 16) {
            const int kk = m / 224;
            const int n  = m - kk*224;
            const int p  = (n >> 4)*18 + (n & 15);
            const int K  = kk*32 + 2*k16;
            const int c0 = ctab[K], c1 = ctab[K+1];
            union { _Float16 h[2]; unsigned u; } pk;
            pk.h[0] = x1[c0 + p];
            pk.h[1] = x1[c1 + p];
            *(unsigned*)(X2f + (m << 5) + 2*k16) = pk.u;   // 4B-aligned b32 store
        }
    }
    __syncthreads();

    // conv2 as im2col GEMM: out(64x224) = W2(64x288) x X2(288x224)
    // Each wave owns an N-tile (16 output pixels) and computes all 4 M-tiles
    // against one B fragment. 9 K-steps of 32; all operand loads are b128.
    const int lg8  = lg * 8;
    const int lg16 = lg * 16;
    #pragma unroll 1
    for (int nt = wv; nt < 14; nt += 8) {
        const int n = nt*16 + ln;           // global output pixel column
        v8f acc[4];
        #pragma unroll
        for (int mt = 0; mt < 4; ++mt)
            acc[mt] = (v8f){0.f,0.f,0.f,0.f,0.f,0.f,0.f,0.f};

        #pragma unroll
        for (int kk = 0; kk < 9; ++kk) {
            const int kbase = kk * 32;
            // B fragment: 16 K-contiguous halves at X2f[(kk*224+n)*32 + lg16]
            const _Float16* xrow = X2f + ((kk*224 + n) << 5) + lg16;
            v8h b0 = *(const v8h*)(xrow);
            v8h b1 = *(const v8h*)(xrow + 8);
            v16h bf = __builtin_shufflevector(b0, b1,
                        0,1,2,3,4,5,6,7, 8,9,10,11,12,13,14,15);
            // 4 M-tiles: A fragment = two contiguous 16B-aligned LDS b128 loads
            #pragma unroll
            for (int mt = 0; mt < 4; ++mt) {
                const _Float16* wrow = &w2h[(mt*16 + ln)*288 + kbase];
                v8h a0 = *(const v8h*)(wrow + lg16);
                v8h a1 = *(const v8h*)(wrow + lg16 + 8);
                v16h a = __builtin_shufflevector(a0, a1,
                            0,1,2,3,4,5,6,7, 8,9,10,11,12,13,14,15);
                acc[mt] = __builtin_amdgcn_wmma_f32_16x16x32_f16(
                    false, a, false, bf, (short)0, acc[mt], false, false);
            }
        }
        // epilogue: bias + relu + spatial-sum into handfeat
        #pragma unroll
        for (int mt = 0; mt < 4; ++mt) {
            #pragma unroll
            for (int r = 0; r < 8; ++r) {
                const int Mg = mt*16 + r + lg8; // C/D layout: VGPR r -> M=r / r+8
                float v = acc[mt][r] + c2b[Mg];
                v = v > 0.f ? v : 0.f;
                atomicAdd(&handfeat[Mg], v);
            }
        }
    }
    __syncthreads();
    if (tid < 64) handfeat[tid] *= (1.0f / 224.0f);   // mean over 14*16
    __syncthreads();

    // ---------------- Phase B: call path + attention + heads ---------------
    float* dkv    = (float*)(smem + 0);
    float* dmask  = (float*)(smem + 10752);
    float* hemb   = (float*)(smem + 11136);
    float* q      = (float*)(smem + 12032);
    float* wlog   = (float*)(smem + 13600);
    float* dq     = (float*)(smem + 18976);
    float* pooled = (float*)(smem + 20544);
    float* cf     = (float*)(smem + 20800);

    // discard K/V: [tile_emb(16) | pos(8) | ply(4)] per (player, slot)
    for (int e = tid; e < 96*28; e += 256) {
        int n = e / 28, d = e - n*28;
        int p = n / 24, s = n - p*24;
        int ix = disc[(b*4 + p)*24 + s]; if (ix < 0) ix = PADIDX;
        float v;
        if (d < 16)       v = temb[ix*16 + d];
        else if (d < 24)  v = pe[s*8 + (d-16)];
        else              v = ye[p*4 + (d-24)];
        dkv[n*28 + d] = v;
        if (d == 0) dmask[n] = (ix != PADIDX) ? 1.f : 0.f;
    }
    // raw hand embedding (for q)
    for (int e = tid; e < 224; e += 256) {
        int i = e >> 4, d = e & 15;
        int ix = hand[b*14 + i]; if (ix < 0) ix = PADIDX;
        hemb[e] = temb[ix*16 + d];
    }
    // pooled calls (masked mean), 4 groups x 16 dims
    if (tid < 64) {
        int g = tid >> 4, j = tid & 15;
        float s = 0.f, cnt = 0.f;
        for (int k = 0; k < 16; ++k) {
            int ix = (g == 0) ? called[b*16 + k] : opp[(b*3 + (g-1))*16 + k];
            if (ix < 0) ix = PADIDX;
            float m = (ix != PADIDX) ? 1.f : 0.f;
            cnt += m;
            s += temb[ix*16 + j] * m;
        }
        pooled[tid] = s / fmaxf(cnt, 1.f);
    }
    __syncthreads();

    // q = hemb @ qpw^T + qpb  (14x28)
    if (tid < 392) {
        int i = tid / 28, d = tid - i*28;
        float s = qpb[d];
        for (int k = 0; k < 16; ++k) s += hemb[i*16 + k] * qpw[d*16 + k];
        q[tid] = s;
    }
    // cf = relu(pooled @ clw^T + clb)  (4x64)
    {
        int g = tid >> 6, o = tid & 63;
        float s = clb[o];
        for (int k = 0; k < 16; ++k) s += pooled[g*16 + k] * clw[o*16 + k];
        cf[tid] = s > 0.f ? s : 0.f;
    }
    __syncthreads();

    // masked logits  (14x96)
    for (int e = tid; e < 14*96; e += 256) {
        int i = e / 96, n = e - i*96;
        float s = 0.f;
        for (int d = 0; d < 28; ++d) s += q[i*28 + d] * dkv[n*28 + d];
        s *= 0.18898223650461363f;  // 1/sqrt(28)
        wlog[e] = (dmask[n] > 0.f) ? s : -1e30f;
    }
    // call_feat = cf @ crw^T + crb -> (4,16) flattened
    if (tid < 64) {
        int g = tid >> 4, j = tid & 15;
        float s = crb[j];
        for (int k = 0; k < 64; ++k) s += cf[g*64 + k] * crw[j*64 + k];
        callfeat[tid] = s;
    }
    __syncthreads();

    // masked softmax per row (serial per-row; rows=14)
    if (tid < 14) {
        float* row = wlog + tid*96;
        float mx = -1e30f;
        for (int n = 0; n < 96; ++n) mx = fmaxf(mx, row[n]);
        float s1 = 0.f;
        for (int n = 0; n < 96; ++n) { float e2 = expf(row[n] - mx); row[n] = e2; s1 += e2; }
        float inv = 1.f / s1;
        float den = 0.f;
        for (int n = 0; n < 96; ++n) { float w = row[n] * inv * dmask[n]; row[n] = w; den += w; }
        float sc = (den > 0.f) ? (1.f / fmaxf(den, 1e-12f)) : 0.f;
        for (int n = 0; n < 96; ++n) row[n] *= sc;
    }
    __syncthreads();

    // dq = w @ dkv  (14x28)
    if (tid < 392) {
        int i = tid / 28, d = tid - i*28;
        float s = 0.f;
        for (int n = 0; n < 96; ++n) s += wlog[i*96 + n] * dkv[n*28 + d];
        dq[tid] = s;
    }
    __syncthreads();

    // d_feat = dq_flat @ qrw^T  (64)
    if (tid < 64) {
        float s = 0.f;
        for (int k = 0; k < 392; ++k) s += dq[k] * qrw[tid*392 + k];
        dfeat[tid] = s;
    }
    __syncthreads();

    // head1: relu([hand|call|dfeat](192) @ h1w^T + h1b)  (128)
    if (tid < 128) {
        float s = h1b[tid];
        const float* wrow = h1w + tid*192;
        for (int k = 0; k < 64; ++k) s += handfeat[k] * wrow[k];
        for (int k = 0; k < 64; ++k) s += callfeat[k] * wrow[64 + k];
        for (int k = 0; k < 64; ++k) s += dfeat[k]    * wrow[128 + k];
        h1s[tid] = s > 0.f ? s : 0.f;
    }
    __syncthreads();

    // head2 -> output (37)
    if (tid < 37) {
        float s = h2b[tid];
        for (int k = 0; k < 128; ++k) s += h1s[k] * h2w[tid*128 + k];
        out[b*37 + tid] = s;
    }
}

extern "C" void kernel_launch(void* const* d_in, const int* in_sizes, int n_in,
                              void* d_out, int out_size, void* d_ws, size_t ws_size,
                              hipStream_t stream) {
    (void)n_in; (void)out_size; (void)d_ws; (void)ws_size;
    const int Bn = in_sizes[0] / 14;
    wallconv_fused_kernel<<<Bn, 256, SMEM_TOTAL, stream>>>(
        (const int*)d_in[0],  (const int*)d_in[1],  (const int*)d_in[2],  (const int*)d_in[3],
        (const float*)d_in[4],  (const float*)d_in[5],  (const float*)d_in[6],
        (const float*)d_in[7],  (const float*)d_in[8],  (const float*)d_in[9],
        (const float*)d_in[10], (const float*)d_in[11], (const float*)d_in[12],
        (const float*)d_in[13], (const float*)d_in[14], (const float*)d_in[15],
        (const float*)d_in[16], (const float*)d_in[17], (const float*)d_in[18],
        (const float*)d_in[19], (const float*)d_in[20], (const float*)d_in[21],
        (float*)d_out);
}
